// VanillaRNN_53979148976508
// MI455X (gfx1250) — matrix operand order repaired and checked by reading, not compiled, not run on hardware
//
#include <hip/hip_runtime.h>

typedef __bf16       v16bf __attribute__((ext_vector_type(16)));
typedef float        v8f   __attribute__((ext_vector_type(8)));
typedef unsigned int v8u   __attribute__((ext_vector_type(8)));

#define RNN_B 512
#define RNN_T 2048
#define RNN_V 32
#define RNN_U 128
#define HPU   68   // padded h-row stride in uints (136 bf16 = 272 B -> bank-friendly, 16B aligned)

static __device__ __forceinline__ unsigned short f2bf(float f) {
    unsigned u = __builtin_bit_cast(unsigned, f);
    unsigned r = u + 0x7fffu + ((u >> 16) & 1u);   // round-to-nearest-even
    return (unsigned short)(r >> 16);
}
static __device__ __forceinline__ unsigned pk2(float a, float b) {
    return (unsigned)f2bf(a) | ((unsigned)f2bf(b) << 16);
}

// CDNA5 async DMA: global memory -> LDS, 16B per lane, tracked by ASYNCcnt.
static __device__ __forceinline__ void async_ld_b128(unsigned lds_off, const void* gaddr) {
    asm volatile("global_load_async_to_lds_b128 %0, %1, off"
                 :: "v"(lds_off), "v"(gaddr) : "memory");
}
static __device__ __forceinline__ void wait_async0() {
    asm volatile("s_wait_asynccnt 0x0" ::: "memory");
}

__global__ __launch_bounds__(256)
void rnn2_wmma_kernel(const float* __restrict__ samples,
                      const float* __restrict__ masks,
                      const float* __restrict__ Wx0, const float* __restrict__ Wh0,
                      const float* __restrict__ b0,
                      const float* __restrict__ Wx1, const float* __restrict__ Wh1,
                      const float* __restrict__ b1,
                      const float* __restrict__ Wout, const float* __restrict__ bout,
                      float* __restrict__ out)
{
    __shared__ float    sb[3][16 * RNN_V];   // samples f32 (scan: argmax + gather)
    __shared__ unsigned sbh[3][16 * 16];     // samples bf16 packed, A-fragment-ready (row stride 16 uints)
    __shared__ float    mb[2][16 * RNN_V];   // mask tiles (filled by async DMA)
    __shared__ unsigned h0b[2][16 * HPU];    // h0 state, bf16 packed, double buffered
    __shared__ unsigned h1b[2][16 * HPU];    // h1 state
    __shared__ float    lg[16 * RNN_V];      // logits (incl. bout, pre-mask)

    const int tid = threadIdx.x;
    const int w   = tid >> 5;      // wave id 0..7 -> owns hidden columns [16w,16w+16)
    const int L   = tid & 31;      // lane
    const int ln  = L & 15;
    const int hi  = L >> 4;
    const int rowbase = blockIdx.x * 16;

    // ---------- build weight B-fragments once (held in VGPRs for all 2048 steps) ----------
    auto loadB = [&](const float* __restrict__ W, int Nt, int kb, int nb) -> v16bf {
        v8u u;
        const int N  = nb * 16 + ln;
        const int k0 = kb * 32 + hi * 16;
        #pragma unroll
        for (int d = 0; d < 8; ++d)
            u[d] = pk2(W[(k0 + 2*d) * Nt + N], W[(k0 + 2*d + 1) * Nt + N]);
        return __builtin_bit_cast(v16bf, u);
    };
    v8u zu = {0u,0u,0u,0u,0u,0u,0u,0u};
    v16bf bWx0f = loadB(Wx0, RNN_U, 0, w);
    v16bf bWh0[4], bWx1[4], bWh1[4], bWoutF[4];
    #pragma unroll
    for (int kb = 0; kb < 4; ++kb) {
        bWh0[kb]   = loadB(Wh0, RNN_U, kb, w);
        bWx1[kb]   = loadB(Wx1, RNN_U, kb, w);
        bWh1[kb]   = loadB(Wh1, RNN_U, kb, w);
        bWoutF[kb] = (w < 2) ? loadB(Wout, RNN_V, kb, w) : __builtin_bit_cast(v16bf, zu);
    }
    const float bias0 = b0[w*16 + ln];
    const float bias1 = b1[w*16 + ln];
    const float biasO = (w < 2) ? bout[w*16 + ln] : 0.f;

    // A 16x32 bf16 fragment from packed-bf16 LDS buffer with row stride `stride` uints
    auto loadA = [&](const unsigned* hb, int kb, int stride) -> v16bf {
        v8u u;
        const int base = ln * stride + kb * 16 + hi * 8;
        #pragma unroll
        for (int d = 0; d < 4; ++d) u[d]     = hb[base + d];
        #pragma unroll
        for (int d = 0; d < 4; ++d) u[d + 4] = hb[base + 4 + d];
        return __builtin_bit_cast(v16bf, u);
    };
    auto mm = [&](v16bf a, v16bf b, v8f c) -> v8f {
        return __builtin_amdgcn_wmma_f32_16x16x32_bf16(false, a, false, b, (short)0, c, false, false);
    };
    // D tile + bias -> ELU (branch-free, native v_exp) -> bf16 -> LDS state
    auto storeH = [&](unsigned* hb, v8f a, float bias) {
        unsigned short* hp = (unsigned short*)hb;
        const int N = w*16 + ln;
        #pragma unroll
        for (int r = 0; r < 8; ++r) {
            float v = a[r] + bias;
            float e = __expf(fminf(v, 0.f)) - 1.f;
            v = (v > 0.f) ? v : e;
            hp[(r + hi*8) * (2*HPU) + N] = f2bf(v);
        }
    };

    // ---------- init: zero state, x_0 = 0, preload s_0 (regs) / m_0 (async DMA) ----------
    for (int i = tid; i < 16*HPU; i += 256) { h0b[0][i] = 0u; h1b[0][i] = 0u; }
    for (int i = tid; i < 16*16; i += 256) sbh[2][i] = 0u;
    {
        const int row = tid >> 4, c = (tid & 15) * 2;
        const long g = (long)(rowbase + row) * (RNN_T * RNN_V) + c;
        float2 sv = *(const float2*)(samples + g);
        *(float2*)&sb[0][row*RNN_V + c] = sv;
        sbh[0][row*16 + (c >> 1)] = pk2(sv.x, sv.y);
    }
    if (w >= 4) {   // waves 4..7: async-DMA mask tile for t=0
        const int row = (w - 4) * 4 + (L >> 3), chunk = L & 7;
        const long g = (long)(rowbase + row) * (RNN_T * RNN_V) + chunk * 4;
        async_ld_b128((unsigned)(unsigned long)&mb[0][row*RNN_V + chunk*4], masks + g);
    }
    wait_async0();
    __syncthreads();

    float acc_lp = 0.f;

    for (int t = 0; t < RNN_T; ++t) {
        const int pr = t & 1, pw = pr ^ 1;

        // async mask DMA issued last step has landed; safe before issuing the next one
        wait_async0();

        // prefetch s_{t+1} (regs -> f32 + bf16 LDS copies), m_{t+1} (async DMA -> LDS)
        if (t + 1 < RNN_T) {
            const int row = tid >> 4, c = (tid & 15) * 2;
            const long g = (long)(rowbase + row) * (RNN_T * RNN_V) + (long)(t + 1) * RNN_V + c;
            float2 sv = *(const float2*)(samples + g);
            *(float2*)&sb[(t + 1) % 3][row*RNN_V + c] = sv;
            sbh[(t + 1) % 3][row*16 + (c >> 1)] = pk2(sv.x, sv.y);
            if (w >= 4) {
                const int mrow = (w - 4) * 4 + (L >> 3), chunk = L & 7;
                const long mg = (long)(rowbase + mrow) * (RNN_T * RNN_V) + (long)(t + 1) * RNN_V + chunk * 4;
                async_ld_b128((unsigned)(unsigned long)&mb[(t + 1) & 1][mrow*RNN_V + chunk*4], masks + mg);
            }
        }

        // -------- layer 0: h0' = elu(x_t @ Wx0 + h0 @ Wh0 + b0) --------
        {
            v16bf xf = loadA(sbh[(t + 2) % 3], 0, 16);       // x_t = s_{t-1}, pre-converted bf16
            v16bf a0 = loadA(h0b[pr], 0, HPU), a1 = loadA(h0b[pr], 1, HPU);
            v16bf a2 = loadA(h0b[pr], 2, HPU), a3 = loadA(h0b[pr], 3, HPU);
            v8f acc = {0.f,0.f,0.f,0.f,0.f,0.f,0.f,0.f};
            acc = mm(xf, bWx0f, acc);
            acc = mm(a0, bWh0[0], acc); acc = mm(a1, bWh0[1], acc);
            acc = mm(a2, bWh0[2], acc); acc = mm(a3, bWh0[3], acc);
            storeH(h0b[pw], acc, bias0);
        }
        __syncthreads();

        // -------- layer 1: h1' = elu(h0' @ Wx1 + h1 @ Wh1 + b1) --------
        {
            v16bf a0 = loadA(h0b[pw], 0, HPU), a1 = loadA(h0b[pw], 1, HPU);
            v16bf a2 = loadA(h0b[pw], 2, HPU), a3 = loadA(h0b[pw], 3, HPU);
            v8f acc = {0.f,0.f,0.f,0.f,0.f,0.f,0.f,0.f};
            acc = mm(a0, bWx1[0], acc); acc = mm(a1, bWx1[1], acc);
            acc = mm(a2, bWx1[2], acc); acc = mm(a3, bWx1[3], acc);
            v16bf c0 = loadA(h1b[pr], 0, HPU), c1 = loadA(h1b[pr], 1, HPU);
            v16bf c2 = loadA(h1b[pr], 2, HPU), c3 = loadA(h1b[pr], 3, HPU);
            acc = mm(c0, bWh1[0], acc); acc = mm(c1, bWh1[1], acc);
            acc = mm(c2, bWh1[2], acc); acc = mm(c3, bWh1[3], acc);
            storeH(h1b[pw], acc, bias1);
        }
        __syncthreads();

        // -------- logits = h1' @ Wout + bout (waves 0,1 cover V=32 columns) --------
        if (w < 2) {
            v16bf a0 = loadA(h1b[pw], 0, HPU), a1 = loadA(h1b[pw], 1, HPU);
            v16bf a2 = loadA(h1b[pw], 2, HPU), a3 = loadA(h1b[pw], 3, HPU);
            v8f la = {0.f,0.f,0.f,0.f,0.f,0.f,0.f,0.f};
            la = mm(a0, bWoutF[0], la); la = mm(a1, bWoutF[1], la);
            la = mm(a2, bWoutF[2], la); la = mm(a3, bWoutF[3], la);
            const int N = w*16 + ln;
            #pragma unroll
            for (int r = 0; r < 8; ++r) lg[(r + hi*8) * RNN_V + N] = la[r] + biasO;
        }
        __syncthreads();

        // -------- log-softmax + gather at argmax(s_t), accumulate (wave 0, 16 lanes) ----
        if (w == 0 && L < 16) {
            const float* lr = lg + L * RNN_V;
            const float* mr = mb[t & 1] + L * RNN_V;
            const float* sr = sb[t % 3] + L * RNN_V;
            float mx = -3.402823466e38f, bs = -3.402823466e38f; int bv = 0;
            for (int v = 0; v < RNN_V; ++v) {
                float l = lr[v] + mr[v];
                mx = fmaxf(mx, l);
                float s = sr[v];
                if (s > bs) { bs = s; bv = v; }
            }
            float sum = 0.f;
            for (int v = 0; v < RNN_V; ++v) sum += __expf(lr[v] + mr[v] - mx);
            acc_lp += (lr[bv] + mr[bv]) - mx - __logf(sum);
        }
        __syncthreads();
    }

    if (w == 0 && L < 16) out[rowbase + L] = acc_lp;
}

extern "C" void kernel_launch(void* const* d_in, const int* in_sizes, int n_in,
                              void* d_out, int out_size, void* d_ws, size_t ws_size,
                              hipStream_t stream) {
    (void)in_sizes; (void)n_in; (void)out_size; (void)d_ws; (void)ws_size;
    const float* samples = (const float*)d_in[0];
    const float* masks   = (const float*)d_in[1];
    const float* Wx0     = (const float*)d_in[2];
    const float* Wh0     = (const float*)d_in[3];
    const float* b0      = (const float*)d_in[4];
    const float* Wx1     = (const float*)d_in[5];
    const float* Wh1     = (const float*)d_in[6];
    const float* b1      = (const float*)d_in[7];
    const float* Wout    = (const float*)d_in[8];
    const float* bout    = (const float*)d_in[9];
    float* out = (float*)d_out;

    dim3 grid(RNN_B / 16), block(256);
    rnn2_wmma_kernel<<<grid, block, 0, stream>>>(samples, masks, Wx0, Wh0, b0,
                                                 Wx1, Wh1, b1, Wout, bout, out);
}
